// P3Embed_1468878815320
// MI455X (gfx1250) — matrix operand order repaired and checked by reading, not compiled, not run on hardware
//
#include <hip/hip_runtime.h>
#include <hip/hip_bf16.h>

typedef _Float16 half_t;
typedef __attribute__((ext_vector_type(16))) _Float16 v16h;
typedef __attribute__((ext_vector_type(8)))  float    v8f;

#define BATCH 8
#define N0    8192
#define M0    2048
#define M1    512
#define KS    32
#define NC0   (BATCH * M0 * KS)   /* 524288 */
#define NC1   (BATCH * M1 * KS)   /* 131072 */
#define R2    0.01f

#define CDIV(a, b) (((a) + (b) - 1) / (b))

// ---------------------------------------------------------------------------
// Furthest point sampling: one block per batch, sequential over M picks.
// ---------------------------------------------------------------------------
__global__ void __launch_bounds__(1024)
fps_kernel(const float* __restrict__ p, float* __restrict__ mind,
           int* __restrict__ idx, int N, int M) {
  int b = blockIdx.x;
  p    += (size_t)b * N * 3;
  mind += (size_t)b * N;
  idx  += (size_t)b * M;
  int t = threadIdx.x, nt = blockDim.x;
  __shared__ float sv[1024];
  __shared__ int   si[1024];
  __shared__ int   sfar;
  for (int i = t; i < N; i += nt) mind[i] = 1e10f;
  if (t == 0) sfar = 0;
  __syncthreads();
  for (int it = 0; it < M; ++it) {
    int far = sfar;
    if (t == 0) idx[it] = far;
    float cx = p[far * 3 + 0], cy = p[far * 3 + 1], cz = p[far * 3 + 2];
    float best = -1.0f;
    int bi = 0;
    for (int i = t; i < N; i += nt) {
      float dx = p[i * 3 + 0] - cx, dy = p[i * 3 + 1] - cy, dz = p[i * 3 + 2] - cz;
      float d  = dx * dx + dy * dy + dz * dz;
      float md = fminf(mind[i], d);
      mind[i] = md;
      if (md > best) { best = md; bi = i; }
    }
    sv[t] = best;
    si[t] = bi;
    __syncthreads();
    for (int s = nt >> 1; s > 0; s >>= 1) {
      if (t < s) {
        if (sv[t + s] > sv[t]) { sv[t] = sv[t + s]; si[t] = si[t + s]; }
      }
      __syncthreads();
    }
    if (t == 0) sfar = si[0];
    __syncthreads();
  }
}

// ---------------------------------------------------------------------------
// Gather sampled centers: cp[b,m,:] = p[b, idx[b,m], :]
// ---------------------------------------------------------------------------
__global__ void gather_cp_kernel(const float* __restrict__ p, const int* __restrict__ idx,
                                 float* __restrict__ cp, int N, int M) {
  int i = blockIdx.x * blockDim.x + threadIdx.x;
  if (i >= BATCH * M) return;
  int b = i / M;
  int n = idx[i];
  const float* s = p + ((size_t)b * N + n) * 3;
  float* d = cp + (size_t)i * 3;
  d[0] = s[0]; d[1] = s[1]; d[2] = s[2];
}

// ---------------------------------------------------------------------------
// Ball query: first KS in-radius points in index order; pad with first hit.
// ---------------------------------------------------------------------------
__global__ void ball_query_kernel(const float* __restrict__ cp, const float* __restrict__ p,
                                  int* __restrict__ nidx, int N, int M) {
  int i = blockIdx.x * blockDim.x + threadIdx.x;
  if (i >= BATCH * M) return;
  int b = i / M;
  const float* c  = cp + (size_t)i * 3;
  const float* pb = p + (size_t)b * N * 3;
  float cx = c[0], cy = c[1], cz = c[2];
  int* o = nidx + (size_t)i * KS;
  int cnt = 0;
  for (int n = 0; n < N && cnt < KS; ++n) {
    float dx = pb[n * 3 + 0] - cx, dy = pb[n * 3 + 1] - cy, dz = pb[n * 3 + 2] - cz;
    if (dx * dx + dy * dy + dz * dz < R2) o[cnt++] = n;
  }
  int f0 = (cnt > 0) ? o[0] : 0;
  for (; cnt < KS; ++cnt) o[cnt] = f0;
}

// ---------------------------------------------------------------------------
// Stage 0 X build: column-major f16 [col][32]; rows 0..2 dp, 3..5 df, pad 0.
// ---------------------------------------------------------------------------
__global__ void build_x0_kernel(const float* __restrict__ p, const float* __restrict__ f,
                                const float* __restrict__ cp, const int* __restrict__ fidx,
                                const int* __restrict__ nidx, half_t* __restrict__ X) {
  int col = blockIdx.x * blockDim.x + threadIdx.x;
  if (col >= NC0) return;
  int b = col / (M0 * KS);
  int r = col % (M0 * KS);
  int m = r / KS;
  int n = nidx[col];
  const float* pn = p + ((size_t)b * N0 + n) * 3;
  const float* cm = cp + ((size_t)b * M0 + m) * 3;
  half_t* x = X + (size_t)col * 32;
  x[0] = (half_t)(pn[0] - cm[0]);
  x[1] = (half_t)(pn[1] - cm[1]);
  x[2] = (half_t)(pn[2] - cm[2]);
  int ci = fidx[b * M0 + m];
#pragma unroll
  for (int c = 0; c < 3; ++c) {
    const float* fc = f + ((size_t)b * 3 + c) * N0;
    x[3 + c] = (half_t)(fc[n] - fc[ci]);
  }
#pragma unroll
  for (int c = 6; c < 32; ++c) x[c] = (half_t)0.0f;
}

// ---------------------------------------------------------------------------
// Stage 1 X build: [col][160]; rows 0..2 dp, 3..130 df (f1 point-major), pad 0.
// ---------------------------------------------------------------------------
__global__ void build_x1_kernel(const float* __restrict__ p, const half_t* __restrict__ f1,
                                const float* __restrict__ cp, const int* __restrict__ fidx,
                                const int* __restrict__ nidx, half_t* __restrict__ X) {
  int col = blockIdx.x * blockDim.x + threadIdx.x;
  if (col >= NC1) return;
  int b = col / (M1 * KS);
  int r = col % (M1 * KS);
  int m = r / KS;
  int n = nidx[col];
  const float* pn = p + ((size_t)b * M0 + n) * 3;   // N1 == M0
  const float* cm = cp + ((size_t)b * M1 + m) * 3;
  half_t* x = X + (size_t)col * 160;
  x[0] = (half_t)(pn[0] - cm[0]);
  x[1] = (half_t)(pn[1] - cm[1]);
  x[2] = (half_t)(pn[2] - cm[2]);
  int ci = fidx[b * M1 + m];
  const half_t* fn = f1 + ((size_t)b * M0 + n) * 128;
  const half_t* fc = f1 + ((size_t)b * M0 + ci) * 128;
  for (int c = 0; c < 128; ++c)
    x[3 + c] = (half_t)((float)fn[c] - (float)fc[c]);
#pragma unroll
  for (int c = 131; c < 160; ++c) x[c] = (half_t)0.0f;
}

// ---------------------------------------------------------------------------
// Weight prep: fold BN into W (f16) + bias (f32), pad cin -> cinp with zeros.
// ---------------------------------------------------------------------------
__global__ void prep_w_bn_kernel(const float* __restrict__ W, const float* __restrict__ gamma,
                                 const float* __restrict__ beta, const float* __restrict__ mean,
                                 const float* __restrict__ var, half_t* __restrict__ Wh,
                                 float* __restrict__ bias, int cout, int cin, int cinp) {
  int i = blockIdx.x * blockDim.x + threadIdx.x;
  if (i >= cout * cinp) return;
  int o = i / cinp, c = i % cinp;
  float s = gamma[o] * rsqrtf(var[o] + 1e-5f);
  Wh[i] = (c < cin) ? (half_t)(W[(size_t)o * cin + c] * s) : (half_t)0.0f;
  if (c == 0) bias[o] = beta[o] - mean[o] * s;
}

__global__ void prep_w_b_kernel(const float* __restrict__ W, const float* __restrict__ b,
                                half_t* __restrict__ Wh, float* __restrict__ bias,
                                int cout, int cin, int cinp) {
  int i = blockIdx.x * blockDim.x + threadIdx.x;
  if (i >= cout * cinp) return;
  int o = i / cinp, c = i % cinp;
  Wh[i] = (c < cin) ? (half_t)W[(size_t)o * cin + c] : (half_t)0.0f;
  if (c == 0) bias[o] = b[o];
}

// ---------------------------------------------------------------------------
// WMMA GEMM: Y[cout x NC] = relu?(W[cout x cinp] * X[cinp x NC] + bias)
//   X/Y column-major (column stride ld). Two K-regions: region A (k < kA)
//   reads srcA at column col>>shA (shA=5 fuses the pooled broadcast),
//   region B reads srcB at column col>>shB.
//   Block: 256 thr = 8 waves, tile 128 rows x 128 cols; 8 wmma/wave/K-step.
//   X tile staged via GLOBAL_LOAD_ASYNC_TO_LDS_B128 into double-buffered LDS;
//   one barrier per K-step.
// ---------------------------------------------------------------------------
__global__ void __launch_bounds__(256)
gemm_wmma(const half_t* __restrict__ Wm, const float* __restrict__ bias,
          const half_t* __restrict__ srcA, int ldA, int kA, int shA,
          const half_t* __restrict__ srcB, int ldB, int kB, int shB,
          half_t* __restrict__ out, int ldo, int relu) {
  const int cinp = kA + kB;
  const int lane = threadIdx.x & 31;
  const int wave = threadIdx.x >> 5;
  const size_t j0 = (size_t)blockIdx.x * 128;
  const int r0 = blockIdx.y * 128;
  const int rr = r0 + wave * 16 + (lane & 15);

  __shared__ half_t xs[2][128 * 40];  // [buf][col][k], stride 40 halves

  v8f acc[8] = {};

  const int kb   = (lane < 16) ? 0 : 8;    // A-frag K base (16x32 f16 layout)
  const int kofs = (lane < 16) ? 0 : 8;    // B-frag K base (same K order as A)
  const int tcol = threadIdx.x >> 1;       // 0..127
  const int hp   = threadIdx.x & 1;        // 16-half segment within column

  union V16 { v16h v; uint4 q[2]; };

  int buf = 0;
  for (int k0 = 0; k0 < cinp; k0 += 32, buf ^= 1) {
    // Async global->LDS stage of 128 cols x 32 k (32B per thread).
    {
      size_t col = j0 + tcol;
      const half_t* s;
      if (k0 < kA) s = srcA + ((col >> shA) * (size_t)ldA) + k0;
      else         s = srcB + ((col >> shB) * (size_t)ldB) + (k0 - kA);
      const void* g = (const void*)(s + hp * 16);
      // LDS byte offset = low 32 bits of the generic pointer (ISA: flat->LDS
      // truncates to addr[31:0], the wave-relative LDS address).
      unsigned ldst = (unsigned)(uintptr_t)(&xs[buf][tcol * 40 + hp * 16]);
      asm volatile(
          "global_load_async_to_lds_b128 %0, %1, off\n\t"
          "global_load_async_to_lds_b128 %0, %1, off offset:16"
          :: "v"(ldst), "v"(g) : "memory");
    }
    asm volatile("s_wait_asynccnt 0x0" ::: "memory");
    __syncthreads();

    // A fragment: 16 rows of W, K = {kb..kb+7, kb+16..kb+23}
    V16 a;
    const half_t* wp = Wm + (size_t)rr * cinp + k0;
    a.q[0] = *(const uint4*)(wp + kb);
    a.q[1] = *(const uint4*)(wp + kb + 16);

    // Batch all 8 B fragments, then 8 back-to-back WMMAs.
    V16 bf[8];
#pragma unroll
    for (int t = 0; t < 8; ++t) {
      const half_t* bp = &xs[buf][(t * 16 + (lane & 15)) * 40 + kofs];
      bf[t].q[0] = *(const uint4*)(bp);
      bf[t].q[1] = *(const uint4*)(bp + 16);
    }
#pragma unroll
    for (int t = 0; t < 8; ++t) {
      acc[t] = __builtin_amdgcn_wmma_f32_16x16x32_f16(
          false, a.v, false, bf[t].v, (short)0, acc[t], false, false);
    }
  }

  // Epilogue: bias + relu, f32 -> f16, contiguous 16B store per tile.
  const int rb = r0 + wave * 16 + ((lane < 16) ? 0 : 8);
  float bv[8];
#pragma unroll
  for (int v = 0; v < 8; ++v) bv[v] = bias[rb + v];
#pragma unroll
  for (int t = 0; t < 8; ++t) {
    size_t cc = j0 + t * 16 + (lane & 15);
    union { uint4 q; half_t h[8]; } o;
#pragma unroll
    for (int v = 0; v < 8; ++v) {
      float y = acc[t][v] + bv[v];
      if (relu) y = fmaxf(y, 0.0f);
      o.h[v] = (half_t)y;
    }
    *(uint4*)(out + cc * (size_t)ldo + rb) = o.q;
  }
}

// ---------------------------------------------------------------------------
// Max-pool over 32-column groups: P[g][c] = max_k Y[(g*32+k)][c]  (f16 -> f16)
// Cw = 1 << cshift.
// ---------------------------------------------------------------------------
__global__ void pool32_kernel(const half_t* __restrict__ Y, half_t* __restrict__ P,
                              int cshift, long long total) {
  long long i = (long long)blockIdx.x * blockDim.x + threadIdx.x;
  if (i >= total) return;
  long long g = i >> cshift;
  int c = (int)(i & ((1 << cshift) - 1));
  const int Cw = 1 << cshift;
  const half_t* base = Y + (g << (cshift + 5)) + c;
  float m = -3.0e38f;
  for (int k = 0; k < 32; ++k) m = fmaxf(m, (float)base[(size_t)k * Cw]);
  P[i] = (half_t)m;
}

// ---------------------------------------------------------------------------
// Final pool: f16 Y [col x 256] -> f32 out [B, 256, M1]
// ---------------------------------------------------------------------------
__global__ void pool_final_kernel(const half_t* __restrict__ Y, float* __restrict__ out) {
  int i = blockIdx.x * blockDim.x + threadIdx.x;
  if (i >= BATCH * 256 * M1) return;
  int b = i / (256 * M1);
  int r = i % (256 * M1);
  int c = r / M1;
  int m = r % M1;
  size_t g = (size_t)b * M1 + m;
  const half_t* base = Y + g * 32 * 256 + c;
  float v = -3.0e38f;
  for (int k = 0; k < 32; ++k) v = fmaxf(v, (float)base[(size_t)k * 256]);
  out[i] = v;
}

// ---------------------------------------------------------------------------
extern "C" void kernel_launch(void* const* d_in, const int* in_sizes, int n_in,
                              void* d_out, int out_size, void* d_ws, size_t ws_size,
                              hipStream_t stream) {
  (void)in_sizes; (void)n_in; (void)out_size; (void)ws_size;
  const float* p = (const float*)d_in[0];
  const float* f = (const float*)d_in[1];

  // Workspace bump allocator (256B aligned). Total ~550 MB.
  char* w = (char*)d_ws;
  auto alloc = [&](size_t bytes) -> char* {
    char* r = w;
    w += (bytes + 255) & ~(size_t)255;
    return r;
  };
  half_t* BufA = (half_t*)alloc((size_t)NC0 * 256 * 2);
  half_t* BufB = (half_t*)alloc((size_t)NC0 * 128 * 2);
  half_t* BufC = (half_t*)alloc((size_t)NC0 * 128 * 2);
  half_t* P    = (half_t*)alloc((size_t)BATCH * M0 * 128 * 2);
  half_t* F1   = (half_t*)alloc((size_t)BATCH * M0 * 128 * 2);
  float*  mind = (float*)alloc((size_t)BATCH * N0 * 4);
  int*    idx0 = (int*)alloc((size_t)BATCH * M0 * 4);
  int*    idx1 = (int*)alloc((size_t)BATCH * M1 * 4);
  int*   nidx0 = (int*)alloc((size_t)NC0 * 4);
  int*   nidx1 = (int*)alloc((size_t)NC1 * 4);
  float*  cp0  = (float*)alloc((size_t)BATCH * M0 * 3 * 4);
  float*  cp1  = (float*)alloc((size_t)BATCH * M1 * 3 * 4);

  // Layers: {s0c1l0, s0c1l1, s0c2l0, s0c2l1, s1c1l0, s1c1l1, s1c2l0, s1c2l1}
  const int couts[8] = {128, 128, 256, 128, 256, 256, 512, 256};
  const int cins[8]  = {  6, 128, 256, 256, 131, 256, 512, 512};
  const int cinps[8] = { 32, 128, 256, 256, 160, 256, 512, 512};
  half_t* Wh[8];
  float*  bs[8];
  for (int l = 0; l < 8; ++l) {
    Wh[l] = (half_t*)alloc((size_t)couts[l] * cinps[l] * 2);
    bs[l] = (float*)alloc((size_t)couts[l] * 4);
  }

  // Param order in d_in: p, f, then per layer in dict insertion order:
  // BN layers: W, gamma, beta, mean, var  |  bias layers: W, b  (None dropped)
  auto prep_bn = [&](int l, int wi) {
    int tot = couts[l] * cinps[l];
    prep_w_bn_kernel<<<CDIV(tot, 256), 256, 0, stream>>>(
        (const float*)d_in[wi], (const float*)d_in[wi + 1], (const float*)d_in[wi + 2],
        (const float*)d_in[wi + 3], (const float*)d_in[wi + 4],
        Wh[l], bs[l], couts[l], cins[l], cinps[l]);
  };
  auto prep_b = [&](int l, int wi) {
    int tot = couts[l] * cinps[l];
    prep_w_b_kernel<<<CDIV(tot, 256), 256, 0, stream>>>(
        (const float*)d_in[wi], (const float*)d_in[wi + 1],
        Wh[l], bs[l], couts[l], cins[l], cinps[l]);
  };
  prep_bn(0, 2);   // s0 conv1 l0: W=2, gamma=3, beta=4, mean=5, var=6
  prep_b (1, 7);   // s0 conv1 l1: W=7, b=8
  prep_bn(2, 9);   // s0 conv2 l0: 9..13
  prep_bn(3, 14);  // s0 conv2 l1: 14..18
  prep_bn(4, 19);  // s1 conv1 l0: 19..23
  prep_b (5, 24);  // s1 conv1 l1: 24,25
  prep_bn(6, 26);  // s1 conv2 l0: 26..30
  prep_bn(7, 31);  // s1 conv2 l1: 31..35

  // ---------------- Stage 0 ----------------
  fps_kernel<<<BATCH, 1024, 0, stream>>>(p, mind, idx0, N0, M0);
  gather_cp_kernel<<<CDIV(BATCH * M0, 256), 256, 0, stream>>>(p, idx0, cp0, N0, M0);
  ball_query_kernel<<<CDIV(BATCH * M0, 256), 256, 0, stream>>>(cp0, p, nidx0, N0, M0);
  build_x0_kernel<<<CDIV(NC0, 256), 256, 0, stream>>>(p, f, cp0, idx0, nidx0, BufA);

  dim3 g0(NC0 / 128, 1);
  dim3 g2(NC0 / 128, 2);
  // conv1 l0: 32 -> 128, bn+relu     (X0=BufA -> Y0=BufB)
  gemm_wmma<<<g0, 256, 0, stream>>>(Wh[0], bs[0], BufA, 32, 32, 0,
                                    nullptr, 0, 0, 0, BufB, 128, 1);
  // conv1 l1: 128 -> 128, bias       (BufB -> Y1=BufC)
  gemm_wmma<<<g0, 256, 0, stream>>>(Wh[1], bs[1], BufB, 128, 128, 0,
                                    nullptr, 0, 0, 0, BufC, 128, 0);
  // pooled = max_k(Y1)
  pool32_kernel<<<CDIV((long long)BATCH * M0 * 128, 256), 256, 0, stream>>>(
      BufC, P, 7, (long long)BATCH * M0 * 128);
  // conv2 l0: [pooled||x] 256 -> 256 (regionA: P broadcast >>5; regionB: Y1)
  gemm_wmma<<<g2, 256, 0, stream>>>(Wh[2], bs[2], P, 128, 128, 5,
                                    BufC, 128, 128, 0, BufA, 256, 1);
  // conv2 l1: 256 -> 128             (BufA -> BufB)
  gemm_wmma<<<g0, 256, 0, stream>>>(Wh[3], bs[3], BufA, 256, 256, 0,
                                    nullptr, 0, 0, 0, BufB, 128, 1);
  // stage0 output features, point-major f16: F1[(b*M0+n)*128 + c]
  pool32_kernel<<<CDIV((long long)BATCH * M0 * 128, 256), 256, 0, stream>>>(
      BufB, F1, 7, (long long)BATCH * M0 * 128);

  // ---------------- Stage 1 ----------------
  fps_kernel<<<BATCH, 1024, 0, stream>>>(cp0, mind, idx1, M0, M1);
  gather_cp_kernel<<<CDIV(BATCH * M1, 256), 256, 0, stream>>>(cp0, idx1, cp1, M0, M1);
  ball_query_kernel<<<CDIV(BATCH * M1, 256), 256, 0, stream>>>(cp1, cp0, nidx1, M0, M1);
  build_x1_kernel<<<CDIV(NC1, 256), 256, 0, stream>>>(cp0, F1, cp1, idx1, nidx1, BufC);

  dim3 s1b(NC1 / 128, 2);
  dim3 s1c(NC1 / 128, 4);
  // conv1 l0: 160 -> 256, bn+relu    (X1=BufC -> BufB)
  gemm_wmma<<<s1b, 256, 0, stream>>>(Wh[4], bs[4], BufC, 160, 160, 0,
                                     nullptr, 0, 0, 0, BufB, 256, 1);
  // conv1 l1: 256 -> 256, bias       (BufB -> BufA)
  gemm_wmma<<<s1b, 256, 0, stream>>>(Wh[5], bs[5], BufB, 256, 256, 0,
                                     nullptr, 0, 0, 0, BufA, 256, 0);
  pool32_kernel<<<CDIV((long long)BATCH * M1 * 256, 256), 256, 0, stream>>>(
      BufA, P, 8, (long long)BATCH * M1 * 256);
  // conv2 l0: [pooled||x] 512 -> 512 (BufA + P -> BufC)
  gemm_wmma<<<s1c, 256, 0, stream>>>(Wh[6], bs[6], P, 256, 256, 5,
                                     BufA, 256, 256, 0, BufC, 512, 1);
  // conv2 l1: 512 -> 256             (BufC -> BufB)
  gemm_wmma<<<s1b, 256, 0, stream>>>(Wh[7], bs[7], BufC, 512, 512, 0,
                                     nullptr, 0, 0, 0, BufB, 256, 1);
  // final max-pool -> f32 d_out [B, 256, M1]
  pool_final_kernel<<<CDIV(BATCH * 256 * M1, 256), 256, 0, stream>>>(
      BufB, (float*)d_out);
}
